// So3kratesInteraction_89910845374838
// MI455X (gfx1250) — matrix operand order repaired and checked by reading, not compiled, not run on hardware
//
#include <hip/hip_runtime.h>

#define DEV __device__ __forceinline__

typedef __attribute__((ext_vector_type(16))) _Float16 v16h;
typedef __attribute__((ext_vector_type(8)))  _Float16 v8h;
typedef __attribute__((ext_vector_type(8)))  float    v8f;

// ---------------- problem constants ----------------
constexpr int NN  = 50000;
constexpr int HH  = 96;
constexpr int EE  = 800000;
constexpr int MT  = 15;     // MTOT
constexpr int NP  = NN * HH;          // halfs per projection matrix in ws

// packed-weight region offsets (in halfs); each 16x16x32 B-tile = 512 halfs
constexpr int PW_QFB  = 0;            // 18 tiles (6 col x 3 k)
constexpr int PW_KFB  = 9216;
constexpr int PW_VFB  = 18432;
constexpr int PW_QGB  = 27648;
constexpr int PW_KGB  = 36864;
constexpr int PW_FBR1 = 46080;        // K=32 -> 6 tiles
constexpr int PW_FBR2 = 49152;
constexpr int PW_FBS2 = 58368;
constexpr int PW_GBR1 = 67584;
constexpr int PW_GBR2 = 70656;
constexpr int PW_GBS2 = 79872;
constexpr int PW_TOT  = 89088;

// LDS-staged copies of the edge-kernel weights (offsets relative to PW_FBR1)
constexpr int LW_FBR1 = 0;
constexpr int LW_FBR2 = 3072;
constexpr int LW_FBS2 = 12288;
constexpr int LW_GBR1 = 21504;
constexpr int LW_GBR2 = 24576;
constexpr int LW_GBS2 = 33792;
constexpr int LW_TOT  = 43008;        // halfs = 84 KB

constexpr int BUF_HALFS = 4 * 4 * 1536;              // per-block tile buffers
constexpr int DYN_HALFS = LW_TOT + BUF_HALFS;        // 67584 halfs
constexpr int DYN_BYTES = DYN_HALFS * 2;             // 135168 B < 320 KB WGP LDS

constexpr float RCUTF = 5.0f;
constexpr float PIF   = 3.14159265358979f;

// ---------------- helpers ----------------
DEV void lds_fence() {
  __builtin_amdgcn_wave_barrier();
  asm volatile("s_wait_dscnt 0x0" ::: "memory");
}

DEV float silu(float z) { return z / (1.0f + __expf(-z)); }

// Pack A-operand (16x32 f16) from f32 row data.
// Lane role: m = lane%16 (caller passes row ptr already offset by kk),
// half = lane/16. i<8 -> k = 8*half+i ; i>=8 -> k = 16+8*half+(i-8).
DEV v16h packA_f32(const float* __restrict__ row, int half) {
  union { v16h v; _Float16 h[16]; } u;
  const float* s1 = row + 8 * half;
  const float* s2 = row + 16 + 8 * half;
#pragma unroll
  for (int i = 0; i < 8; ++i) {
    u.h[i]     = (_Float16)s1[i];
    u.h[8 + i] = (_Float16)s2[i];
  }
  return u.v;
}

// Same A packing but from f16 LDS data (two ds_load_b128 per chunk).
DEV v16h packA_f16lds(const _Float16* rowkk, int half) {
  union { v16h v; v8h p[2]; } u;
  u.p[0] = *(const v8h*)(rowkk + 8 * half);
  u.p[1] = *(const v8h*)(rowkk + 16 + 8 * half);
  return u.v;
}

// three-way dot: packed f16 product (v_pk_mul_f16) then mixed-precision FMA accum
DEV float dot8(v8h a, v8h b, v8h c) {
  v8h t = a * b;
  float s = 0.f;
#pragma unroll
  for (int i = 0; i < 8; ++i) s = fmaf((float)t[i], (float)c[i], s);
  return s;
}

// ---------------- kernel: pre-pack a weight matrix into WMMA B layout ----------------
// W is row-major (K x 96) f32. One wave per (ct,kc) tile; lane writes its 16 halfs.
// B layout: lane half hf, elem i<8 -> K=8*hf+i, i>=8 -> K=16+8*hf+(i-8) of the 32-chunk;
// n = ct*16 + lane%16.
__global__ void k_pack_w(const float* __restrict__ W, _Float16* __restrict__ dst, int nkc) {
  const int t    = blockIdx.x;           // t = ct*nkc + kc
  const int kc   = t % nkc;
  const int ct   = t / nkc;
  const int lane = threadIdx.x;
  const int nl   = lane & 15;
  const int hf   = lane >> 4;
  _Float16* o = dst + t * 512 + lane * 16;
  const int n = ct * 16 + nl;
#pragma unroll
  for (int i = 0; i < 16; ++i)
    o[i] = (_Float16)W[(kc * 32 + (i < 8 ? 8 * hf + i : 16 + 8 * hf + (i - 8))) * 96 + n];
}

// ---------------- kernel: node projections x@W for 5 weight mats ----------------
// grid (N/16, 5), block 192 (6 waves); wave w computes col tile w of matrix blockIdx.y
__global__ __launch_bounds__(192) void k_node_proj(const float* __restrict__ x,
                                                   const _Float16* __restrict__ pw,
                                                   _Float16* __restrict__ proj) {
  __shared__ float xs[16 * 96];
  const int nodeBase = blockIdx.x * 16;
  const int mat      = blockIdx.y;
  for (int i = threadIdx.x; i < 16 * 96; i += 192)
    xs[i] = x[(size_t)nodeBase * 96 + i];
  __syncthreads();

  const int wave = threadIdx.x >> 5;
  const int lane = threadIdx.x & 31;
  const int nl   = lane & 15;
  const int half = lane >> 4;

  v8f c = {};
#pragma unroll
  for (int kc = 0; kc < 3; ++kc) {
    v16h a = packA_f32(xs + nl * 96 + kc * 32, half);
    v16h b = *(const v16h*)(pw + mat * 9216 + (wave * 3 + kc) * 512 + lane * 16);
    c = __builtin_amdgcn_wmma_f32_16x16x32_f16(false, a, false, b, (short)0, c, false, false);
  }
  _Float16* dm = proj + (size_t)mat * NP;
#pragma unroll
  for (int r = 0; r < 8; ++r)
    dm[(size_t)(nodeBase + r + 8 * half) * 96 + wave * 16 + nl] = (_Float16)c[r];
}

// ---------------- per-tile fused 2-layer filter MLP (rad WMMA + sph VALU + layer2 WMMA) ----------------
// B operands now come from the LDS-staged weight block `lw` (ds_load_b128 pairs).
DEV void mlp_block(const float* __restrict__ ea, int e0,
                   const _Float16* lw, int offR1, int offR2, int offS2,
                   const float* __restrict__ s_w1, const float* __restrict__ s_b1,
                   const float* __restrict__ r_b1, const float* __restrict__ r_b2,
                   const float* __restrict__ s_b2,
                   float d0, float d1, float d2,
                   _Float16* hR, _Float16* hS, _Float16* wOut,
                   int nl, int half, int lane) {
  // layer-1 sph branch (K=3): pure VALU, lane covers 48 channels of its edge row
  for (int cl = 0; cl < 48; ++cl) {
    const int c = 48 * half + cl;
    float z = fmaf(d0, s_w1[c], fmaf(d1, s_w1[96 + c], fmaf(d2, s_w1[192 + c], s_b1[c])));
    hS[nl * 96 + c] = (_Float16)silu(z);
  }
  // layer-1 radial branch: (16x32)@(32x96) via 6 WMMAs
  v16h aea = packA_f32(ea + (size_t)(e0 + nl) * 32, half);
#pragma unroll
  for (int ct = 0; ct < 6; ++ct) {
    v8f c = {};
    v16h b = *(const v16h*)(lw + offR1 + ct * 512 + lane * 16);
    c = __builtin_amdgcn_wmma_f32_16x16x32_f16(false, aea, false, b, (short)0, c, false, false);
    const float bias = r_b1[ct * 16 + nl];
#pragma unroll
    for (int r = 0; r < 8; ++r)
      hR[(r + 8 * half) * 96 + ct * 16 + nl] = (_Float16)silu(c[r] + bias);
  }
  lds_fence();
  // layer-2: w = hR@w2r + hS@w2s + b2r + b2s  (K = 96 + 96)
#pragma unroll
  for (int ct = 0; ct < 6; ++ct) {
    v8f c = {};
#pragma unroll
    for (int kc = 0; kc < 3; ++kc) {
      v16h a = packA_f16lds(hR + nl * 96 + kc * 32, half);
      v16h b = *(const v16h*)(lw + offR2 + (ct * 3 + kc) * 512 + lane * 16);
      c = __builtin_amdgcn_wmma_f32_16x16x32_f16(false, a, false, b, (short)0, c, false, false);
    }
#pragma unroll
    for (int kc = 0; kc < 3; ++kc) {
      v16h a = packA_f16lds(hS + nl * 96 + kc * 32, half);
      v16h b = *(const v16h*)(lw + offS2 + (ct * 3 + kc) * 512 + lane * 16);
      c = __builtin_amdgcn_wmma_f32_16x16x32_f16(false, a, false, b, (short)0, c, false, false);
    }
    const float bias = r_b2[ct * 16 + nl] + s_b2[ct * 16 + nl];
#pragma unroll
    for (int r = 0; r < 8; ++r)
      wOut[(r + 8 * half) * 96 + ct * 16 + nl] = (_Float16)(c[r] + bias);
  }
  lds_fence();
}

// ---------------- main edge kernel: MLPs + attention + scatter ----------------
// block = 128 threads = 4 waves; wave handles 32 edges (2 subtiles of 16)
// dynamic LDS: [0,43008) staged packed weights; then 4 waves x {hR,hS,wF,wG} x 1536
__global__ __launch_bounds__(128, 1) void k_edge(
    const float* __restrict__ ea, const float* __restrict__ chi,
    const float* __restrict__ sph, const float* __restrict__ ewt,
    const int* __restrict__ eidx,
    const _Float16* __restrict__ qf, const _Float16* __restrict__ kf,
    const _Float16* __restrict__ vf, const _Float16* __restrict__ qg,
    const _Float16* __restrict__ kg, const _Float16* __restrict__ pw,
    const float* __restrict__ fbs_w1, const float* __restrict__ fbs_b1,
    const float* __restrict__ fbr_b1, const float* __restrict__ fbr_b2,
    const float* __restrict__ fbs_b2,
    const float* __restrict__ gbs_w1, const float* __restrict__ gbs_b1,
    const float* __restrict__ gbr_b1, const float* __restrict__ gbr_b2,
    const float* __restrict__ gbs_b2,
    float* __restrict__ out_x, float* __restrict__ out_chi) {
  extern __shared__ v8h dynv[];
  _Float16* dyn = (_Float16*)dynv;

  const int tid    = threadIdx.x;
  const int waveId = tid >> 5;
  const int lane   = tid & 31;
  const int nl     = lane & 15;
  const int half   = lane >> 4;

  // stage all edge-kernel packed weights (84 KB) into LDS once per block
  {
    const v8h* g = (const v8h*)(pw + PW_FBR1);
    v8h* l = (v8h*)dyn;
    for (int i = tid; i < LW_TOT / 8; i += 128) l[i] = g[i];
  }
  __syncthreads();

  const _Float16* lw = dyn;
  _Float16* bufs = dyn + LW_TOT;
  _Float16* hR = bufs + (waveId * 4 + 0) * 1536;
  _Float16* hS = bufs + (waveId * 4 + 1) * 1536;
  _Float16* wF = bufs + (waveId * 4 + 2) * 1536;
  _Float16* wG = bufs + (waveId * 4 + 3) * 1536;

  const int ew0 = blockIdx.x * 128 + waveId * 32;

  for (int sub = 0; sub < 2; ++sub) {
    const int e   = ew0 + sub * 16 + nl;      // both halves share the edge
    const int e0  = ew0 + sub * 16;
    const int src = eidx[e];
    const int dst = eidx[EE + e];

    if (sub == 0) {  // prefetch next subtile's streaming rows (global_prefetch_b8)
      __builtin_prefetch(ea + (size_t)(e0 + 16 + nl) * 32, 0, 1);
      __builtin_prefetch(sph + (size_t)(e + 16) * MT, 0, 1);
    }

    // L0 CG contraction of chi differences -> d_chi[3] (both halves redundant)
    float d0 = 0.f, d1 = 0.f, d2 = 0.f;
#pragma unroll
    for (int t = 0; t < MT; ++t) {
      const float d  = chi[src * MT + t] - chi[dst * MT + t];
      const float wc = (t < 3) ? 0.57735027f : ((t < 8) ? 0.44721360f : 0.37796447f);
      const float v  = d * d * wc;
      if (t < 3) d0 += v; else if (t < 8) d1 += v; else d2 += v;
    }

    // feature-block filter, then geometric-block filter (reusing h buffers)
    mlp_block(ea, e0, lw, LW_FBR1, LW_FBR2, LW_FBS2,
              fbs_w1, fbs_b1, fbr_b1, fbr_b2, fbs_b2,
              d0, d1, d2, hR, hS, wF, nl, half, lane);
    mlp_block(ea, e0, lw, LW_GBR1, LW_GBR2, LW_GBS2,
              gbs_w1, gbs_b1, gbr_b1, gbr_b2, gbs_b2,
              d0, d1, d2, hR, hS, wG, nl, half, lane);

    // cosine cutoff
    const float w_e = ewt[e];
    const float Ce  = (w_e < RCUTF) ? 0.5f * (__cosf(PIF * w_e * (1.0f / RCUTF)) + 1.0f) : 0.0f;

    const size_t db = (size_t)dst * 96 + 48 * half;
    const size_t sb = (size_t)src * 96 + 48 * half;
    const _Float16* qdp = qf + db;
    const _Float16* kfp = kf + sb;
    const _Float16* vfp = vf + sb;
    const _Float16* qgp = qg + db;
    const _Float16* kgp = kg + sb;
    const _Float16* wfp = wF + nl * 96 + 48 * half;
    const _Float16* wgp = wG + nl * 96 + 48 * half;

    // ---- feature block attention: heads are 24 wide; each half owns 2 whole heads
    float accA = 0.f, accB = 0.f;
#pragma unroll
    for (int ch = 0; ch < 6; ++ch) {
      v8h q = *(const v8h*)(qdp + ch * 8);
      v8h k = *(const v8h*)(kfp + ch * 8);
      v8h w = *(const v8h*)(wfp + ch * 8);
      const float s = dot8(q, w, k);
      if (ch < 3) accA += s; else accB += s;
    }
    const float sf = Ce * 0.20412414f;     // 1/sqrt(24)
    accA *= sf; accB *= sf;
#pragma unroll
    for (int ch = 0; ch < 6; ++ch) {
      v8h v = *(const v8h*)(vfp + ch * 8);
      const float al = (ch < 3) ? accA : accB;
#pragma unroll
      for (int j = 0; j < 8; ++j)
        atomicAdd(out_x + db + ch * 8 + j, al * (float)v[j]);
    }

    // ---- geometric block attention: heads 32 wide; head1 straddles halves
    float g0 = 0.f, g1 = 0.f, g2 = 0.f;
#pragma unroll
    for (int ch = 0; ch < 6; ++ch) {
      v8h q = *(const v8h*)(qgp + ch * 8);
      v8h k = *(const v8h*)(kgp + ch * 8);
      v8h w = *(const v8h*)(wgp + ch * 8);
      const float s    = dot8(q, w, k);
      const int   head = (48 * half + ch * 8) >> 5;
      if (head == 0) g0 += s; else if (head == 1) g1 += s; else g2 += s;
    }
    g0 += __shfl_xor(g0, 16, 32);
    g1 += __shfl_xor(g1, 16, 32);
    g2 += __shfl_xor(g2, 16, 32);
    const float sg  = Ce * 0.17677670f;    // 1/sqrt(32)
    const float ag0 = g0 * sg, ag1 = g1 * sg, ag2 = g2 * sg;

    const int j0 = half ? 8 : 0;
    const int j1 = half ? 15 : 8;
    for (int j = j0; j < j1; ++j) {
      const float a = (j < 3) ? ag0 : ((j < 8) ? ag1 : ag2);
      atomicAdd(out_chi + (size_t)dst * MT + j, a * sph[(size_t)e * MT + j]);
    }
    lds_fence();
  }
}

// ---------------- host entry ----------------
extern "C" void kernel_launch(void* const* d_in, const int* in_sizes, int n_in,
                              void* d_out, int out_size, void* d_ws, size_t ws_size,
                              hipStream_t stream) {
  (void)in_sizes; (void)n_in;
  const float* x      = (const float*)d_in[0];
  const float* chi    = (const float*)d_in[1];
  const float* ea     = (const float*)d_in[2];
  const float* sph    = (const float*)d_in[3];
  const float* ewt    = (const float*)d_in[4];
  const int*   eidx   = (const int*)d_in[5];
  const float* wq_fb  = (const float*)d_in[6];
  const float* wk_fb  = (const float*)d_in[7];
  const float* wv_fb  = (const float*)d_in[8];
  const float* wq_gb  = (const float*)d_in[9];
  const float* wk_gb  = (const float*)d_in[10];
  const float* fbr_w1 = (const float*)d_in[11];
  const float* fbr_b1 = (const float*)d_in[12];
  const float* fbr_w2 = (const float*)d_in[13];
  const float* fbr_b2 = (const float*)d_in[14];
  const float* fbs_w1 = (const float*)d_in[15];
  const float* fbs_b1 = (const float*)d_in[16];
  const float* fbs_w2 = (const float*)d_in[17];
  const float* fbs_b2 = (const float*)d_in[18];
  const float* gbr_w1 = (const float*)d_in[19];
  const float* gbr_b1 = (const float*)d_in[20];
  const float* gbr_w2 = (const float*)d_in[21];
  const float* gbr_b2 = (const float*)d_in[22];
  const float* gbs_w1 = (const float*)d_in[23];
  const float* gbs_b1 = (const float*)d_in[24];
  const float* gbs_w2 = (const float*)d_in[25];
  const float* gbs_b2 = (const float*)d_in[26];

  float* out_x   = (float*)d_out;
  float* out_chi = out_x + (size_t)NN * HH;

  _Float16* wsh = (_Float16*)d_ws;
  _Float16* pw  = wsh + (size_t)5 * NP;
  const size_t need = ((size_t)5 * NP + PW_TOT) * sizeof(_Float16);
  if (ws_size < need) return;

  // allow 132 KB of dynamic LDS for the fused edge kernel (WGP has 320 KB)
  (void)hipFuncSetAttribute((const void*)k_edge,
                            hipFuncAttributeMaxDynamicSharedMemorySize, DYN_BYTES);

  // outputs are accumulated with atomics -> zero every call
  hipMemsetAsync(d_out, 0, (size_t)out_size * sizeof(float), stream);

  // pre-pack weights into WMMA B layout (f16)
  k_pack_w<<<18, 32, 0, stream>>>(wq_fb,  pw + PW_QFB,  3);
  k_pack_w<<<18, 32, 0, stream>>>(wk_fb,  pw + PW_KFB,  3);
  k_pack_w<<<18, 32, 0, stream>>>(wv_fb,  pw + PW_VFB,  3);
  k_pack_w<<<18, 32, 0, stream>>>(wq_gb,  pw + PW_QGB,  3);
  k_pack_w<<<18, 32, 0, stream>>>(wk_gb,  pw + PW_KGB,  3);
  k_pack_w<<< 6, 32, 0, stream>>>(fbr_w1, pw + PW_FBR1, 1);
  k_pack_w<<<18, 32, 0, stream>>>(fbr_w2, pw + PW_FBR2, 3);
  k_pack_w<<<18, 32, 0, stream>>>(fbs_w2, pw + PW_FBS2, 3);
  k_pack_w<<< 6, 32, 0, stream>>>(gbr_w1, pw + PW_GBR1, 1);
  k_pack_w<<<18, 32, 0, stream>>>(gbr_w2, pw + PW_GBR2, 3);
  k_pack_w<<<18, 32, 0, stream>>>(gbs_w2, pw + PW_GBS2, 3);

  // node projections q/k/v (fb) + q/k (gb) -> f16 in workspace (L2-resident)
  k_node_proj<<<dim3(NN / 16, 5), 192, 0, stream>>>(x, pw, wsh);

  // fused edge pipeline
  k_edge<<<EE / 128, 128, DYN_BYTES, stream>>>(
      ea, chi, sph, ewt, eidx,
      wsh + 0 * (size_t)NP, wsh + 1 * (size_t)NP, wsh + 2 * (size_t)NP,
      wsh + 3 * (size_t)NP, wsh + 4 * (size_t)NP, pw,
      fbs_w1, fbs_b1, fbr_b1, fbr_b2, fbs_b2,
      gbs_w1, gbs_b1, gbr_b1, gbr_b2, gbs_b2,
      out_x, out_chi);
}
